// MultiHeadResonanceAttention_12128987644460
// MI455X (gfx1250) — matrix operand order repaired and checked by reading, not compiled, not run on hardware
//
#include <hip/hip_runtime.h>

typedef __attribute__((ext_vector_type(16))) _Float16 v16h;
typedef __attribute__((ext_vector_type(8)))  float    v8f;
typedef __attribute__((ext_vector_type(4)))  unsigned int u32x4;
typedef __attribute__((ext_vector_type(8)))  int      i32x8;
typedef __attribute__((ext_vector_type(4)))  int      i32x4;

static constexpr int Bn = 2, Tn = 2048, Cn = 1024, Hn = 16, HDn = 64;
static constexpr int LDQKV = 3 * Cn;  // 3072

// ---------------- WMMA helpers (CDNA5 v_wmma_f32_16x16x32_f16) ----------------

__device__ __forceinline__ v8f wmma_f16(v16h a, v16h b, v8f c) {
  return __builtin_amdgcn_wmma_f32_16x16x32_f16(false, a, false, b, (short)0, c,
                                                false, false);
}

// A fragment, 16x32 f16, row-major [16][K]; lane L (row=L&15):
// halves 0..15 -> K=koff..koff+15, koff=(L>=16)?16:0.
__device__ __forceinline__ v16h load_frag_a(const _Float16* tile, int ld) {
  const int lane = threadIdx.x & 31;
  const int row  = lane & 15;
  const int koff = (lane & 16) ? 16 : 0;
  const _Float16* p = tile + (long)row * ld + koff;
  union { v16h v; uint4 q[2]; } u;
  u.q[0] = *reinterpret_cast<const uint4*>(p);
  u.q[1] = *reinterpret_cast<const uint4*>(p + 8);
  return u.v;
}

// B fragment for A@B^T, B row-major [N][K]; lane L (col=L&15): K=koff+i, koff=(L>=16)?16:0.
__device__ __forceinline__ v16h load_frag_bT(const _Float16* tile, int ld) {
  const int lane = threadIdx.x & 31;
  const int col  = lane & 15;
  const int koff = (lane & 16) ? 16 : 0;
  const _Float16* p = tile + (long)col * ld + koff;
  union { v16h v; uint4 q[2]; } u;
  u.q[0] = *reinterpret_cast<const uint4*>(p);
  u.q[1] = *reinterpret_cast<const uint4*>(p + 8);
  return u.v;
}

// B fragment with B stored k-major: element (k,n) at tile[k*ld + n].
__device__ __forceinline__ v16h load_frag_b_kmajor(const _Float16* tile, int ld) {
  const int lane = threadIdx.x & 31;
  const int col  = lane & 15;
  const int koff = (lane & 16) ? 16 : 0;
  v16h b;
#pragma unroll
  for (int i = 0; i < 16; ++i) b[i] = tile[(long)(koff + i) * ld + col];
  return b;
}

__device__ __forceinline__ float red_max16(float x) {
  x = fmaxf(x, __shfl_xor(x, 1, 32));
  x = fmaxf(x, __shfl_xor(x, 2, 32));
  x = fmaxf(x, __shfl_xor(x, 4, 32));
  x = fmaxf(x, __shfl_xor(x, 8, 32));
  return x;
}
__device__ __forceinline__ float red_sum16(float x) {
  x += __shfl_xor(x, 1, 32);
  x += __shfl_xor(x, 2, 32);
  x += __shfl_xor(x, 4, 32);
  x += __shfl_xor(x, 8, 32);
  return x;
}

__device__ __forceinline__ unsigned lds_off(const void* p) {
  // generic pointer to LDS: low 32 bits are the LDS byte address (ISA 10.2 aperture rule)
  return (unsigned)(size_t)p;
}

// Issue one TDM 2-D tile load: rows x width f16, global row stride `stride` (halves).
// D# per ISA 8.3/8.4: group0 = {count=1, lds_addr, global_addr, type=2}; group1 holds
// dims/strides; trailing groups zero (<=2-D tensor, tile_dim2=0).
__device__ __forceinline__ void tdm_load_2d(unsigned lds_addr, const _Float16* gptr,
                                            unsigned rows, unsigned width,
                                            unsigned stride) {
  const unsigned long long ga = (unsigned long long)(size_t)gptr;
  u32x4 g0;
  g0[0] = 1u;                                            // count=1 (valid), user mode
  g0[1] = lds_addr;                                      // LDS byte address
  g0[2] = (unsigned)(ga & 0xFFFFFFFFu);                  // global_addr[31:0]
  g0[3] = (unsigned)((ga >> 32) & 0x01FFFFFFu) | (2u << 30);  // addr[56:32] | type=2
  i32x8 g1;
  g1[0] = (int)(1u << 16);                               // data_size=1 (2 bytes)
  g1[1] = (int)((width & 0xFFFFu) << 16);                // tensor_dim0[15:0]
  g1[2] = (int)((width >> 16) | ((rows & 0xFFFFu) << 16)); // dim0 hi | tensor_dim1 lo
  g1[3] = (int)((rows >> 16) | ((width & 0xFFFFu) << 16)); // dim1 hi | tile_dim0
  g1[4] = (int)(rows & 0xFFFFu);                         // tile_dim1 (tile_dim2=0)
  g1[5] = (int)stride;                                   // tensor_dim0_stride[31:0]
  g1[6] = 0;                                             // stride hi / dim1_stride lo
  g1[7] = 0;
  const i32x4 gz4 = {0, 0, 0, 0};
  const i32x8 gz8 = {0, 0, 0, 0, 0, 0, 0, 0};
  // clang-23 / therock-10.0 form: (g0, g1, g2, g3, extra, cpol)
  __builtin_amdgcn_tensor_load_to_lds(g0, g1, gz4, gz4, gz8, 0);
}

// ---------------- Kernels ----------------

__global__ void cvt_f32_to_f16(const float* __restrict__ in,
                               _Float16* __restrict__ out, long n) {
  long i = (long)blockIdx.x * blockDim.x + threadIdx.x;
  const long stride = (long)gridDim.x * blockDim.x;
  for (; i < n; i += stride) out[i] = (_Float16)in[i];
}

// C[M][N] = A[M][K] @ B[N][K]^T ; 64x64 tile per 128-thread block, 4 waves.
// The 64x32 B tile (shared by all 4 waves) is staged in LDS via async-to-LDS
// copies, double-buffered; A fragments come straight from global (per-wave).
__global__ void __launch_bounds__(128)
gemm_abT_wmma(const _Float16* __restrict__ A, const _Float16* __restrict__ Bw,
              float* __restrict__ Cf, _Float16* __restrict__ Ch,
              int K, int lda, int ldb, int ldc) {
  __shared__ __align__(16) _Float16 Bs[2][64 * 32];  // [n][k], 4 KB per buffer

  const int t    = threadIdx.x;
  const int lane = t & 31;
  const int wave = t >> 5;
  const long m0 = (long)blockIdx.y * 64 + wave * 16;
  const long n0 = (long)blockIdx.x * 64;

  const v8f zero = {0.f, 0.f, 0.f, 0.f, 0.f, 0.f, 0.f, 0.f};
  v8f acc[4] = {zero, zero, zero, zero};
  const _Float16* Ap = A + m0 * lda;

  auto stage = [&](int buf, int k) {
    // 64 rows x 64 bytes = 256 16-byte chunks; 2 per thread
#pragma unroll
    for (int c = 0; c < 2; ++c) {
      const int chunk = t * 2 + c;          // 0..255
      const int row   = chunk >> 2;         // output-column row 0..63
      const int part  = chunk & 3;          // 16B sub-chunk
      const _Float16* g = Bw + (n0 + row) * (long)ldb + k + part * 8;
      const unsigned l = lds_off(&Bs[buf][0]) + (unsigned)chunk * 16u;
      asm volatile("global_load_async_to_lds_b128 %0, %1, off"
                   :: "v"(l), "v"(g) : "memory");
    }
  };

  const int nk = K / 32;
  stage(0, 0);
  for (int kt = 0; kt < nk; ++kt) {
    const int k = kt * 32;
    asm volatile("s_wait_asynccnt 0" ::: "memory");  // my async writes landed
    __syncthreads();                                 // tile visible to all waves
    if (kt + 1 < nk) stage((kt + 1) & 1, k + 32);    // overlap next tile
    __builtin_prefetch(Ap + k + 128, 0, 0);          // global_prefetch_b8 (A stream)
    v16h a = load_frag_a(Ap + k, lda);
    const _Float16* Bsh = &Bs[kt & 1][0];
#pragma unroll
    for (int nt = 0; nt < 4; ++nt) {
      v16h b = load_frag_bT(Bsh + nt * 16 * 32, 32);  // ds_load_b128 pair
      acc[nt] = wmma_f16(a, b, acc[nt]);
    }
  }

  const int col   = lane & 15;
  const int rbase = (lane & 16) ? 8 : 0;
#pragma unroll
  for (int nt = 0; nt < 4; ++nt) {
#pragma unroll
    for (int r = 0; r < 8; ++r) {
      const long idx = (m0 + rbase + r) * ldc + n0 + nt * 16 + col;
      if (Cf) Cf[idx] = acc[nt][r];
      else    Ch[idx] = (_Float16)acc[nt][r];
    }
  }
}

// Flash attention over qkv [B*T][3C] f16; head h at cols h*64 (+C for K, +2C for V).
// grid = (B*H, T/64); 4 waves, wave owns 16 queries. K/V 32x64 tiles are DMA'd
// into LDS by the Tensor Data Mover (wave 0), double-buffered; the k-tile loop is
// uniform across the block so barriers are legal.
__global__ void __launch_bounds__(128)
flash_attn_wmma(const _Float16* __restrict__ qkv, _Float16* __restrict__ y) {
  __shared__ __align__(16) _Float16 Kbuf[2][32 * 64];  // [key][d]
  __shared__ __align__(16) _Float16 Vbuf[2][32 * 64];  // [key][d]
  __shared__ __align__(16) _Float16 Pl[4][16 * 32];    // per-wave P staging

  const int lane  = threadIdx.x & 31;
  const int wave  = threadIdx.x >> 5;
  const int b     = blockIdx.x / Hn;
  const int h     = blockIdx.x % Hn;
  const int q0    = blockIdx.y * 64 + wave * 16;
  const int rbase = (lane & 16) ? 8 : 0;
  const int col   = lane & 15;

  const _Float16* base = qkv + (long)b * Tn * LDQKV + h * HDn;
  const _Float16* Qp = base + (long)q0 * LDQKV;
  const _Float16* Kg = base + Cn;
  const _Float16* Vg = base + 2 * Cn;

  const v16h aQ0 = load_frag_a(Qp, LDQKV);
  const v16h aQ1 = load_frag_a(Qp + 32, LDQKV);

  const v8f zero = {0.f, 0.f, 0.f, 0.f, 0.f, 0.f, 0.f, 0.f};
  v8f acc[4] = {zero, zero, zero, zero};
  float mrow[8], lrow[8];
#pragma unroll
  for (int r = 0; r < 8; ++r) { mrow[r] = -3.0e38f; lrow[r] = 0.f; }

  _Float16* Pw = &Pl[wave][0];
  const int ktMax = (blockIdx.y * 64 + 48) / 32 + 1;  // causal bound of last wave

  if (wave == 0) {  // TDM: one DMA per tile, issued once per wave
    tdm_load_2d(lds_off(&Kbuf[0][0]), Kg, 32, 64, LDQKV);
    tdm_load_2d(lds_off(&Vbuf[0][0]), Vg, 32, 64, LDQKV);
  }

  for (int kt = 0; kt < ktMax; ++kt) {
    const int kBase = kt * 32;
    if (wave == 0) __builtin_amdgcn_s_wait_tensorcnt(0);
    __syncthreads();  // K/V tile visible to all waves
    if (wave == 0 && kt + 1 < ktMax) {
      const long nb = (long)(kBase + 32) * LDQKV;
      tdm_load_2d(lds_off(&Kbuf[(kt + 1) & 1][0]), Kg + nb, 32, 64, LDQKV);
      tdm_load_2d(lds_off(&Vbuf[(kt + 1) & 1][0]), Vg + nb, 32, 64, LDQKV);
    }

    if (kBase <= q0 + 15) {  // this wave still has causal work in this tile
      const _Float16* Ksh = &Kbuf[kt & 1][0];
      const _Float16* Vsh = &Vbuf[kt & 1][0];

      v8f s[2] = {zero, zero};
#pragma unroll
      for (int nt = 0; nt < 2; ++nt) {
        v16h b0 = load_frag_bT(Ksh + nt * 16 * 64, 64);
        v16h b1 = load_frag_bT(Ksh + nt * 16 * 64 + 32, 64);
        s[nt] = wmma_f16(aQ0, b0, s[nt]);
        s[nt] = wmma_f16(aQ1, b1, s[nt]);
      }

#pragma unroll
      for (int nt = 0; nt < 2; ++nt) {
#pragma unroll
        for (int r = 0; r < 8; ++r) {
          const int q = q0 + rbase + r;
          const int k = kBase + nt * 16 + col;
          s[nt][r] = (q >= k) ? s[nt][r] * 0.125f : -3.0e38f;
        }
      }

#pragma unroll
      for (int r = 0; r < 8; ++r) {
        float mx   = red_max16(fmaxf(s[0][r], s[1][r]));
        float mnew = fmaxf(mrow[r], mx);
        float corr = __expf(mrow[r] - mnew);
        float p0   = __expf(s[0][r] - mnew);
        float p1   = __expf(s[1][r] - mnew);
        lrow[r] = lrow[r] * corr + red_sum16(p0 + p1);
        mrow[r] = mnew;
#pragma unroll
        for (int nt = 0; nt < 4; ++nt) acc[nt][r] *= corr;
        Pw[(rbase + r) * 32 + col]      = (_Float16)p0;
        Pw[(rbase + r) * 32 + col + 16] = (_Float16)p1;
      }
      asm volatile("s_wait_dscnt 0" ::: "memory");  // per-wave LDS RAW on P

      v16h aP = load_frag_a(Pw, 32);
#pragma unroll
      for (int nt = 0; nt < 4; ++nt) {
        v16h bV = load_frag_b_kmajor(Vsh + nt * 16, 64);
        acc[nt] = wmma_f16(aP, bV, acc[nt]);
      }
    }
  }

#pragma unroll
  for (int r = 0; r < 8; ++r) {
    const float inv = 1.0f / lrow[r];
    const long row = (long)b * Tn + q0 + rbase + r;
#pragma unroll
    for (int nt = 0; nt < 4; ++nt) {
      y[row * Cn + h * HDn + nt * 16 + col] = (_Float16)(acc[nt][r] * inv);
    }
  }
}

// ---------------- Host launcher ----------------

extern "C" void kernel_launch(void* const* d_in, const int* in_sizes, int n_in,
                              void* d_out, int out_size, void* d_ws, size_t ws_size,
                              hipStream_t stream) {
  const float* x     = (const float*)d_in[0];
  const float* w_qkv = (const float*)d_in[1];
  const float* w_o   = (const float*)d_in[2];
  float* out = (float*)d_out;

  const long nx   = (long)Bn * Tn * Cn;
  const long nwq  = 3L * Cn * Cn;
  const long nwo  = (long)Cn * Cn;
  const long nqkv = (long)Bn * Tn * 3 * Cn;

  _Float16* xh   = (_Float16*)d_ws;
  _Float16* wqh  = xh  + nx;
  _Float16* woh  = wqh + nwq;
  _Float16* qkvh = woh + nwo;
  _Float16* yh   = qkvh + nqkv;  // ~48 MB f16 scratch total

  cvt_f32_to_f16<<<2048, 256, 0, stream>>>(x,     xh,  nx);
  cvt_f32_to_f16<<<2048, 256, 0, stream>>>(w_qkv, wqh, nwq);
  cvt_f32_to_f16<<<1024, 256, 0, stream>>>(w_o,   woh, nwo);

  {  // QKV = x @ w_qkv^T : M=4096, N=3072, K=1024 -> f16
    dim3 grid(3 * Cn / 64, Bn * Tn / 64);
    gemm_abT_wmma<<<grid, 128, 0, stream>>>(xh, wqh, nullptr, qkvh,
                                            Cn, Cn, Cn, 3 * Cn);
  }
  {  // attention
    dim3 grid(Bn * Hn, Tn / 64);
    flash_attn_wmma<<<grid, 128, 0, stream>>>(qkvh, yh);
  }
  {  // out = y @ w_o^T : M=4096, N=1024, K=1024 -> f32
    dim3 grid(Cn / 64, Bn * Tn / 64);
    gemm_abT_wmma<<<grid, 128, 0, stream>>>(yh, woh, out, nullptr,
                                            Cn, Cn, Cn, Cn);
  }
}
